// AIM_8985071583610
// MI455X (gfx1250) — matrix operand order repaired and checked by reading, not compiled, not run on hardware
//
#include <hip/hip_runtime.h>
#include <hip/hip_bf16.h>
#include <math.h>

// Problem constants (match reference)
#define B_SZ  128
#define IN_SZ 256
#define HID   512
#define NU    512
#define TOPKN 32
#define KSZ   64
#define QSZ   64
#define VSZ   400

typedef __attribute__((ext_vector_type(16))) __bf16 v16bf;
typedef __attribute__((ext_vector_type(8)))  float  v8f;
typedef unsigned int u32x4 __attribute__((ext_vector_type(4)));
typedef unsigned int u32x8 __attribute__((ext_vector_type(8)));

__device__ __forceinline__ __bf16 f2bf(float f) { return (__bf16)f; }

// ---------------------------------------------------------------------------
// TDM issue: DMA a 2-D tile W[u, 0:256, v0:v0+16] (f32, row stride 400 elems)
// into LDS at byte offset `laddr`, with 1-dword padding after each 16-dword
// row (17-dword LDS row stride -> bank-conflict-free B reads).
// D# built per CDNA5 ISA; 2-group (2-D) form.
// ---------------------------------------------------------------------------
__device__ __forceinline__ void tdm_issue_wtile(const float* W, int u, int v0,
                                                unsigned laddr) {
  const unsigned uu = (unsigned)__builtin_amdgcn_readfirstlane(u);
  const unsigned la = (unsigned)__builtin_amdgcn_readfirstlane((int)laddr);
  const unsigned long long gaddr =
      (unsigned long long)(size_t)W +
      ((unsigned long long)uu * (IN_SZ * VSZ) + (unsigned)v0) * 4ull;
  const unsigned glo = (unsigned)__builtin_amdgcn_readfirstlane((int)(unsigned)gaddr);
  const unsigned ghi = (unsigned)__builtin_amdgcn_readfirstlane((int)(unsigned)(gaddr >> 32));

  u32x4 g0;
  g0[0] = 1u;                       // count=1, user mode
  g0[1] = la;                       // lds_addr (bytes)
  g0[2] = glo;                      // global_addr[31:0]
  g0[3] = (ghi & 0x01FFFFFFu)       // global_addr[56:32]
        | (2u << 30);               // type=2 ("image")

  u32x8 g1;
  g1[0] = (2u << 16)                // data_size = 4 bytes
        | (1u << 20)                // pad_enable
        | (3u << 22)                // pad_interval: 16 dwords
        | (0u << 25);               // pad_amount: 1 dword
  g1[1] = (16u << 16);              // tensor_dim0 = 16 (atomic_barrier_addr=0)
  g1[2] = (256u << 16);             // tensor_dim1 = 256 (tensor_dim0 hi = 0)
  g1[3] = (16u << 16);              // tile_dim0 = 16 (tensor_dim1 hi = 0)
  g1[4] = 256u;                     // tile_dim1 = 256, tile_dim2 = 0
  g1[5] = 400u;                     // tensor_dim0_stride = 400
  g1[6] = 0u;                       // stride hi / dim1_stride lo
  g1[7] = 0u;                       // dim1_stride hi

  asm volatile("tensor_load_to_lds %0, %1" :: "s"(g0), "s"(g1) : "memory");
}

// ---------------------------------------------------------------------------
// Kernel 1: query_layer[u, q] = sum_h hs[u,h] * query_w[u,h,q]
// One block per unit u; streams query_w (64 MB) once, fully coalesced in q.
// ---------------------------------------------------------------------------
__global__ void AIM_query_kernel(const float* __restrict__ hs,
                                 const float* __restrict__ qw,
                                 float* __restrict__ q) {
  __shared__ float hss[HID];
  __shared__ float pq[256];
  const int u = blockIdx.x;
  for (int h = threadIdx.x; h < HID; h += 256)
    hss[h] = hs[(size_t)u * HID + h];
  __syncthreads();
  const int qc  = threadIdx.x & 63;   // output column
  const int seg = threadIdx.x >> 6;   // h-segment (4 segs of 128)
  const float* qwu = qw + (size_t)u * HID * QSZ;
  float acc = 0.f;
  for (int hh = 0; hh < 128; ++hh) {
    const int h = seg * 128 + hh;
    acc += hss[h] * qwu[(size_t)h * QSZ + qc];
    if ((hh & 31) == 0)
      __builtin_prefetch(qwu + (size_t)(h + 64) * QSZ + qc, 0, 1);
  }
  pq[threadIdx.x] = acc;
  __syncthreads();
  if (threadIdx.x < 64) {
    q[(size_t)u * QSZ + threadIdx.x] =
        pq[threadIdx.x] + pq[64 + threadIdx.x] +
        pq[128 + threadIdx.x] + pq[192 + threadIdx.x];
  }
}

// ---------------------------------------------------------------------------
// Kernel 2: kb[b, q] = key_b[q] + sum_i x[b,i] * key_w[q,i]    (tiny)
// ---------------------------------------------------------------------------
__global__ void AIM_key_kernel(const float* __restrict__ x,
                               const float* __restrict__ kw,
                               const float* __restrict__ kbias,
                               float* __restrict__ kb) {
  __shared__ float xs[IN_SZ];
  const int b = blockIdx.x;
  for (int i = threadIdx.x; i < IN_SZ; i += 64)
    xs[i] = x[(size_t)b * IN_SZ + i];
  __syncthreads();
  const int qc = threadIdx.x;
  float acc = kbias[qc];
  const float* kwr = kw + (size_t)qc * IN_SZ;
  #pragma unroll 8
  for (int i = 0; i < IN_SZ; ++i) acc += xs[i] * kwr[i];
  kb[(size_t)b * KSZ + qc] = acc;
}

// ---------------------------------------------------------------------------
// Kernel 3: att[b,u] = sigmoid( (q[u]·kb[b]) / 8 )
// (2-way softmax vs a zero logit == sigmoid; key_b==0, x2 row1 == 0)
// ---------------------------------------------------------------------------
__global__ void AIM_score_kernel(const float* __restrict__ q,
                                 const float* __restrict__ kb,
                                 float* __restrict__ att) {
  __shared__ float qu[QSZ];
  const int u = blockIdx.x;
  if (threadIdx.x < QSZ) qu[threadIdx.x] = q[(size_t)u * QSZ + threadIdx.x];
  __syncthreads();
  const int b = threadIdx.x;          // 128 threads = 128 batch rows
  const float* kbp = kb + (size_t)b * KSZ;
  float s = 0.f;
  #pragma unroll 8
  for (int k = 0; k < KSZ; ++k) s += qu[k] * kbp[k];
  s *= 0.125f;                        // 1/sqrt(64)
  att[(size_t)b * NU + u] = 1.0f / (1.0f + __expf(-s));
}

// ---------------------------------------------------------------------------
// Kernel 4: exact top-32 per batch via pairwise rank (ties -> lower index,
// matching jax.lax.top_k). w[b,u] = att[b,u] if selected else 0.
// ---------------------------------------------------------------------------
__global__ void AIM_topk_kernel(const float* __restrict__ att,
                                float* __restrict__ w) {
  __shared__ float s[NU];
  const int b = blockIdx.x;
  const int u = threadIdx.x;
  s[u] = att[(size_t)b * NU + u];
  __syncthreads();
  const float vu = s[u];
  int rank = 0;
  for (int j = 0; j < NU; ++j) {
    const float vj = s[j];
    rank += (vj > vu) || (vj == vu && j < u);
  }
  w[(size_t)b * NU + u] = (rank < TOPKN) ? vu : 0.0f;
}

// ---------------------------------------------------------------------------
// Kernel 5: out[b,v] += sum_u w[b,u] * sum_i x[b,i]*W[u,i,v]  via bf16 WMMA.
// Grid (25 v-tiles, 8 b-tiles, 16 u-slabs of 32); block = 128 thr = 4 waves.
// Per active unit u: wave 0 TDM-loads W[u,:,vtile] (16 KB, padded) into a
// double-buffered LDS tile; all 4 waves split the K=256 range (64 rows each,
// 2 WMMAs) and accumulate the same 16x16 f32 tile. TENSORcnt + s_barrier
// pipeline the DMA against the matrix math. LDS reduce, then f32 atomics.
// ---------------------------------------------------------------------------
__global__ void AIM_value_kernel(const float* __restrict__ x,
                                 const float* __restrict__ w,
                                 const float* __restrict__ W,
                                 float* __restrict__ out) {
  __shared__ float xs[16][IN_SZ];        // 16 KB: x tile
  __shared__ float wt[16][32];           //  2 KB: masked-att tile
  __shared__ int   alist[32];            // compacted active units
  __shared__ int   acount_s;
  __shared__ float Wbuf[2][256 * 17];    // 2 x 17 KB: TDM tiles (17-dw rows)
  __shared__ float part[4][256];         //  4 KB: per-wave partial C tiles

  const int v0 = blockIdx.x * 16;
  const int b0 = blockIdx.y * 16;
  const int u0 = blockIdx.z * 32;
  const int tid = threadIdx.x;

  for (int idx = tid; idx < 16 * IN_SZ; idx += 128) {
    const int m = idx >> 8, i = idx & 255;
    xs[m][i] = x[(size_t)(b0 + m) * IN_SZ + i];
  }
  for (int idx = tid; idx < 16 * 32; idx += 128) {
    const int m = idx >> 5, j = idx & 31;
    wt[m][j] = w[(size_t)(b0 + m) * NU + u0 + j];
  }
  __syncthreads();
  if (tid == 0) {
    int cnt = 0;
    for (int j = 0; j < 32; ++j) {
      int any = 0;
      for (int m = 0; m < 16; ++m) any |= (wt[m][j] != 0.0f);
      if (any) alist[cnt++] = j;
    }
    acount_s = cnt;
  }
  __syncthreads();
  const int acount = acount_s;

  const int wid  = tid >> 5;
  const int lane = tid & 31;
  const int half = lane >> 4;   // 0: lanes 0-15, 1: lanes 16-31
  const int mn   = lane & 15;   // A row (batch) / B,C column (value)

  const unsigned lds0 = (unsigned)(size_t)&Wbuf[0][0];
  const unsigned lds1 = (unsigned)(size_t)&Wbuf[1][0];

  // Preamble: prefetch first active unit's W tile.
  if (wid == 0 && acount > 0)
    tdm_issue_wtile(W, u0 + alist[0], v0, lds0);

  v8f c = {0.f, 0.f, 0.f, 0.f, 0.f, 0.f, 0.f, 0.f};

  for (int j = 0; j < acount; ++j) {
    const int cur = j & 1;
    if (wid == 0) {
      if (j + 1 < acount) {
        tdm_issue_wtile(W, u0 + alist[j + 1], v0, cur ? lds0 : lds1);
        __builtin_amdgcn_s_wait_tensorcnt(1);   // buf[cur] complete
      } else {
        __builtin_amdgcn_s_wait_tensorcnt(0);
      }
    }
    __syncthreads();                            // publish buf[cur] to all waves

    const int uj = alist[j];
    const float wbu = wt[mn][uj];
    const float* bufc = &Wbuf[cur][0];
    const int ibase = wid * 64;                 // this wave's K slice
    #pragma unroll
    for (int cch = 0; cch < 2; ++cch) {
      const int ic = ibase + cch * 32;
      v16bf a, bm;
      // A operand, contiguous k order used here (same as B): lanes 0-15 M=0-15; elem e -> K = e+16*half
      #pragma unroll
      for (int e = 0; e < 16; ++e) {
        const int kA = e + (half << 4);
        a[e] = f2bf(wbu * xs[mn][ic + kA]);
      }
      // B operand, contiguous k order used here (same as A): lanes 0-15 K=0-15, lanes 16-31 K=16-31; N = mn
      #pragma unroll
      for (int e = 0; e < 16; ++e) {
        const int row = ic + e + (half << 4);
        bm[e] = f2bf(bufc[row * 17 + mn]);      // 17-dw padded rows: no bank conflict
      }
      c = __builtin_amdgcn_wmma_f32_16x16x32_bf16(
              /*neg_a=*/false, a, /*neg_b=*/false, bm,
              /*c_mod=*/(short)0, c, /*reuse_a=*/false, /*reuse_b=*/false);
    }
    __syncthreads();                            // done reading buf[cur]
  }

  // C/D layout: VGPR r holds M = r + 8*half, N = mn
  #pragma unroll
  for (int r = 0; r < 8; ++r) {
    const int m = r + (half << 3);
    part[wid][m * 16 + mn] = c[r];
  }
  __syncthreads();
  for (int e2 = tid; e2 < 256; e2 += 128) {
    const float s = part[0][e2] + part[1][e2] + part[2][e2] + part[3][e2];
    const int m = e2 >> 4, n = e2 & 15;
    atomicAdd(&out[(size_t)(b0 + m) * VSZ + v0 + n], s);
  }
}

__global__ void AIM_zero_kernel(float* __restrict__ out, int n) {
  const int i = blockIdx.x * blockDim.x + threadIdx.x;
  if (i < n) out[i] = 0.0f;
}

// ---------------------------------------------------------------------------
extern "C" void kernel_launch(void* const* d_in, const int* in_sizes, int n_in,
                              void* d_out, int out_size, void* d_ws, size_t ws_size,
                              hipStream_t stream) {
  const float* x     = (const float*)d_in[0];  // (128, 1, 256)
  const float* hs    = (const float*)d_in[1];  // (512, 512)
  const float* key_w = (const float*)d_in[2];  // (64, 256)
  const float* key_b = (const float*)d_in[3];  // (64,)
  const float* hvw   = (const float*)d_in[4];  // (512, 256, 400)
  const float* qw    = (const float*)d_in[5];  // (512, 512, 64)
  float* out = (float*)d_out;                  // (128, 400)

  float* ws  = (float*)d_ws;
  float* q   = ws;                    // 512*64
  float* kb  = q  + NU * QSZ;         // 128*64
  float* att = kb + B_SZ * KSZ;       // 128*512
  float* w   = att + B_SZ * NU;       // 128*512

  AIM_query_kernel<<<NU, 256, 0, stream>>>(hs, qw, q);
  AIM_key_kernel<<<B_SZ, 64, 0, stream>>>(x, key_w, key_b, kb);
  AIM_score_kernel<<<NU, 128, 0, stream>>>(q, kb, att);
  AIM_topk_kernel<<<B_SZ, NU, 0, stream>>>(att, w);
  AIM_zero_kernel<<<(B_SZ * VSZ + 255) / 256, 256, 0, stream>>>(out, B_SZ * VSZ);
  dim3 g5(VSZ / 16, B_SZ / 16, NU / 32);
  AIM_value_kernel<<<g5, 128, 0, stream>>>(x, w, hvw, out);
}
